// Hbnn_47278999994743
// MI455X (gfx1250) — matrix operand order repaired and checked by reading, not compiled, not run on hardware
//
#include <hip/hip_runtime.h>
#include <hip/hip_bf16.h>
#include <math.h>

typedef __attribute__((ext_vector_type(16))) _Float16 v16h;
typedef __attribute__((ext_vector_type(8)))  float    v8f;

#define B_    512
#define IN_   456
#define H_    400
#define OUT_  10
#define GMAX  4
#define MC_   100
#define KT1   15   // ceil(456/32) k-tiles for layer 1 (pad K to 480)
#define KT3   13   // ceil(400/32) k-tiles for layer 2 (pad K to 416)
#define BT    32   // 512/16 b-tiles
#define HT    25   // 400/16 h-tiles

// ---- CDNA5 WMMA VGPR layouts (ISA 7.12.2, wave32) --------------------------
// 16-bit A-matrix 16x32 (MxK): lane L, half j (0..15):
//   M = L%16 ; K = (j&7) + 8*(L/16) + 16*(j>>3)
//   => per lane: two contiguous 8-element K-runs at c0 and c0+16, c0 = 8*(L/16)
__device__ __forceinline__ void a16_layout(int lane, int j, int& m, int& k) {
    m = lane & 15;
    k = (j & 7) + ((lane >> 4) << 3) + ((j >> 3) << 4);
}
// 16-bit B-matrix 32x16 (KxN): lane L, half j (0..15): N = L%16 ; K = (j&7) + 8*(L/16) + 16*(j>>3)
__device__ __forceinline__ void b16_layout(int lane, int j, int& k, int& n) {
    n = lane & 15;
    k = (j & 7) + ((lane >> 4) << 3) + ((j >> 3) << 4);
}
// f32 C/D 16x16: lane L, vgpr r: N = L%16 ; M = r + 8*(L/16)

__device__ __forceinline__ float softplus2(float s) {
    float sp = (s > 20.0f) ? s : log1pf(__expf(s));
    return sp * sp;
}

// ---------------------------------------------------------------------------
__global__ void zero_out_kernel(float* out, int n) {
    int i = blockIdx.x * blockDim.x + threadIdx.x;
    if (i < n) out[i] = 0.0f;
}

// Pack x and x*x into A-matrix layout: xA[btile][kt][lane][16] halves.
__global__ void pack_x_kernel(const float* __restrict__ x,
                              _Float16* __restrict__ xA,
                              _Float16* __restrict__ x2A) {
    int idx = blockIdx.x * blockDim.x + threadIdx.x;       // total = BT*KT1*32*16
    if (idx >= BT * KT1 * 32 * 16) return;
    int j     = idx & 15;
    int lane  = (idx >> 4) & 31;
    int kt    = (idx >> 9) % KT1;
    int btile = idx / (512 * KT1);
    int m, k;  a16_layout(lane, j, m, k);
    int b   = btile * 16 + m;
    int col = kt * 32 + k;
    float v = (col < IN_) ? x[(size_t)b * IN_ + col] : 0.0f;
    xA[idx]  = (_Float16)v;
    x2A[idx] = (_Float16)(v * v);
}

// Pack mu1 / softplus(sigma1)^2 into B layout: w[g][kt][htile][lane][16].
__global__ void pack_w1_kernel(const float* __restrict__ mu1,
                               const float* __restrict__ sigma1,
                               _Float16* __restrict__ w1mu,
                               _Float16* __restrict__ w1sig) {
    int idx = blockIdx.x * blockDim.x + threadIdx.x;       // total = G*KT1*HT*32*16
    if (idx >= GMAX * KT1 * HT * 32 * 16) return;
    int j    = idx & 15;
    int lane = (idx >> 4) & 31;
    int ht   = (idx >> 9) % HT;
    int kt   = (idx / (512 * HT)) % KT1;
    int g    = idx / (512 * HT * KT1);
    int k, n;  b16_layout(lane, j, k, n);
    int col = kt * 32 + k;       // input-feature index, pad >= 456 -> 0
    int h   = ht * 16 + n;       // hidden index, always < 400
    float m = 0.0f, s2 = 0.0f;
    if (col < IN_) {
        size_t off = ((size_t)g * IN_ + col) * H_ + h;
        m  = mu1[off];
        s2 = softplus2(sigma1[off]);
    }
    w1mu[idx]  = (_Float16)m;
    w1sig[idx] = (_Float16)s2;
}

// Pack mu3 / softplus(sigma3)^2 into B layout: w[g][kt][lane][16] (N padded 10->16).
__global__ void pack_w3_kernel(const float* __restrict__ mu3,
                               const float* __restrict__ sigma3,
                               _Float16* __restrict__ w3mu,
                               _Float16* __restrict__ w3sig) {
    int idx = blockIdx.x * blockDim.x + threadIdx.x;       // total = G*KT3*32*16
    if (idx >= GMAX * KT3 * 32 * 16) return;
    int j    = idx & 15;
    int lane = (idx >> 4) & 31;
    int kt   = (idx >> 9) % KT3;
    int g    = idx / (512 * KT3);
    int k, n;  b16_layout(lane, j, k, n);
    int col = kt * 32 + k;       // hidden index, pad >= 400 -> 0
    float m = 0.0f, s2 = 0.0f;
    if (col < H_ && n < OUT_) {
        size_t off = ((size_t)g * H_ + col) * OUT_ + n;
        m  = mu3[off];
        s2 = softplus2(sigma3[off]);
    }
    w3mu[idx]  = (_Float16)m;
    w3sig[idx] = (_Float16)s2;
}

// Layer 1: G1 = x @ mu1 ; SD1 = sqrt(x^2 @ sp2(sigma1)).  One wave per 16x16 tile.
__global__ void layer1_wmma_kernel(const _Float16* __restrict__ xA,
                                   const _Float16* __restrict__ x2A,
                                   const _Float16* __restrict__ w1mu,
                                   const _Float16* __restrict__ w1sig,
                                   float* __restrict__ G1,
                                   float* __restrict__ SD1,
                                   const int* __restrict__ nump) {
    int btile = blockIdx.x;      // 0..31
    int htile = blockIdx.y;      // 0..24
    int g     = blockIdx.z;      // 0..3
    if (g > *nump) return;       // wave-uniform: EXEC stays all-ones for WMMA
    int lane  = threadIdx.x;     // 0..31

    v8f cmu = {};
    v8f csg = {};
    #pragma unroll 1
    for (int kt = 0; kt < KT1; ++kt) {
        const v16h a  = *(const v16h*)(xA  + ((((size_t)btile * KT1 + kt) * 32 + lane) << 4));
        const v16h a2 = *(const v16h*)(x2A + ((((size_t)btile * KT1 + kt) * 32 + lane) << 4));
        size_t woff = (((((size_t)g * KT1 + kt) * HT + htile) * 32 + lane) << 4);
        const v16h bm = *(const v16h*)(w1mu  + woff);
        const v16h bs = *(const v16h*)(w1sig + woff);
        cmu = __builtin_amdgcn_wmma_f32_16x16x32_f16(false, a,  false, bm, (short)0, cmu, false, false);
        csg = __builtin_amdgcn_wmma_f32_16x16x32_f16(false, a2, false, bs, (short)0, csg, false, false);
    }

    int n     = lane & 15;
    int group = lane >> 4;
    int hcol  = htile * 16 + n;
    #pragma unroll
    for (int r = 0; r < 8; ++r) {
        int b = btile * 16 + r + (group << 3);
        size_t o = ((size_t)g * B_ + b) * H_ + hcol;
        G1[o]  = cmu[r];
        SD1[o] = sqrtf(fmaxf(csg[r], 0.0f));
    }
}

// Layer 2: per (g, mc, 16-row b-tile): h = relu(G1 + SD1*z1) built in-register
// from b128-vectorized zeta1 / LDS reads, two f16 WMMAs contract h and h^2
// against mu3/sp2(sigma3) over K=400, then relu -> softmax -> atomic MC-mean.
#define L2PAD 416
__global__ void layer2_wmma_kernel(const float* __restrict__ G1,
                                   const float* __restrict__ SD1,
                                   const _Float16* __restrict__ w3mu,
                                   const _Float16* __restrict__ w3sig,
                                   const float* __restrict__ zeta1,
                                   const float* __restrict__ zeta3,
                                   float* __restrict__ out,
                                   const int* __restrict__ nump) {
    __shared__ float sG[16 * L2PAD];
    __shared__ float sS[16 * L2PAD];

    int btile = blockIdx.x;              // 0..31
    int mcc   = blockIdx.y;              // 0..24
    int g     = blockIdx.z;              // 0..3
    if (g > *nump) return;
    int tid   = threadIdx.x;             // 0..127 (4 waves)
    int lane  = tid & 31;
    int wave  = tid >> 5;
    int mc    = mcc * 4 + wave;

    // Stage this b-tile's G1 / sqrt(D1) rows into LDS once, float4 at a time
    // (row stride 1600B and L2PAD stride 1664B are both 16B multiples).
    {
        float4* sG4 = (float4*)sG;
        float4* sS4 = (float4*)sS;
        const int Q = L2PAD / 4;                 // 104 float4 per row
        for (int i = tid; i < 16 * Q; i += 128) {
            int row = i / Q, col = (i % Q) * 4;
            float4 gv = make_float4(0.f, 0.f, 0.f, 0.f);
            float4 sv = gv;
            if (col < H_) {
                size_t o = ((size_t)g * B_ + btile * 16 + row) * H_ + col;
                gv = *(const float4*)(G1 + o);
                sv = *(const float4*)(SD1 + o);
            }
            sG4[i] = gv;
            sS4[i] = sv;
        }
    }
    __syncthreads();

    const int m     = lane & 15;
    const int group = lane >> 4;
    const float* zb   = zeta1 + (((size_t)g * MC_ + mc) * B_ + btile * 16 + m) * H_;
    const float* gRow = sG + m * L2PAD;
    const float* sRow = sS + m * L2PAD;

    v8f cmu = {};
    v8f csg = {};

    // Main k-tiles 0..11: all 32 columns in-range; lane's 16 K-values are two
    // aligned 32B runs -> 4x global_load_b128 + 8x ds_load_b128 per tile.
    #pragma unroll 1
    for (int kt = 0; kt < KT3 - 1; ++kt) {
        const int c0 = kt * 32 + (group << 3);   // run0: c0..c0+7, run1: c0+16..c0+23
        float zr[16] __attribute__((aligned(16)));
        float gr[16] __attribute__((aligned(16)));
        float sr[16] __attribute__((aligned(16)));
        *(float4*)(zr + 0)  = *(const float4*)(zb + c0);
        *(float4*)(zr + 4)  = *(const float4*)(zb + c0 + 4);
        *(float4*)(zr + 8)  = *(const float4*)(zb + c0 + 16);
        *(float4*)(zr + 12) = *(const float4*)(zb + c0 + 20);
        __builtin_prefetch(zb + c0 + 32, 0, 0);  // next k-tile of the zeta1 stream
        *(float4*)(gr + 0)  = *(const float4*)(gRow + c0);
        *(float4*)(gr + 4)  = *(const float4*)(gRow + c0 + 4);
        *(float4*)(gr + 8)  = *(const float4*)(gRow + c0 + 16);
        *(float4*)(gr + 12) = *(const float4*)(gRow + c0 + 20);
        *(float4*)(sr + 0)  = *(const float4*)(sRow + c0);
        *(float4*)(sr + 4)  = *(const float4*)(sRow + c0 + 4);
        *(float4*)(sr + 8)  = *(const float4*)(sRow + c0 + 16);
        *(float4*)(sr + 12) = *(const float4*)(sRow + c0 + 20);

        v16h a, a2;
        #pragma unroll
        for (int j = 0; j < 16; ++j) {
            float hv = fmaxf(gr[j] + sr[j] * zr[j], 0.0f);
            a[j]  = (_Float16)hv;
            a2[j] = (_Float16)(hv * hv);
        }
        size_t woff = ((((size_t)g * KT3 + kt) * 32 + lane) << 4);
        const v16h bm = *(const v16h*)(w3mu  + woff);
        const v16h bs = *(const v16h*)(w3sig + woff);
        cmu = __builtin_amdgcn_wmma_f32_16x16x32_f16(false, a,  false, bm, (short)0, cmu, false, false);
        csg = __builtin_amdgcn_wmma_f32_16x16x32_f16(false, a2, false, bs, (short)0, csg, false, false);
    }

    // Tail k-tile (kt=12): run0 (cols 384+8g..391+8g) is always < 400; run1
    // (cols >= 400) is entirely padding -> zero, no per-element guards needed.
    {
        const int c0 = (KT3 - 1) * 32 + (group << 3);
        float zr[8] __attribute__((aligned(16)));
        float gr[8] __attribute__((aligned(16)));
        float sr[8] __attribute__((aligned(16)));
        *(float4*)(zr + 0) = *(const float4*)(zb + c0);
        *(float4*)(zr + 4) = *(const float4*)(zb + c0 + 4);
        *(float4*)(gr + 0) = *(const float4*)(gRow + c0);
        *(float4*)(gr + 4) = *(const float4*)(gRow + c0 + 4);
        *(float4*)(sr + 0) = *(const float4*)(sRow + c0);
        *(float4*)(sr + 4) = *(const float4*)(sRow + c0 + 4);

        v16h a = {}, a2 = {};
        #pragma unroll
        for (int j = 0; j < 8; ++j) {
            float hv = fmaxf(gr[j] + sr[j] * zr[j], 0.0f);
            a[j]  = (_Float16)hv;
            a2[j] = (_Float16)(hv * hv);
        }
        size_t woff = ((((size_t)g * KT3 + (KT3 - 1)) * 32 + lane) << 4);
        const v16h bm = *(const v16h*)(w3mu  + woff);
        const v16h bs = *(const v16h*)(w3sig + woff);
        cmu = __builtin_amdgcn_wmma_f32_16x16x32_f16(false, a,  false, bm, (short)0, cmu, false, false);
        csg = __builtin_amdgcn_wmma_f32_16x16x32_f16(false, a2, false, bs, (short)0, csg, false, false);
    }

    // Epilogue: C layout -> row M = r + 8*group, class N = lane%16 (valid N<10).
    const int  n     = lane & 15;
    const bool valid = (n < OUT_);
    #pragma unroll 1
    for (int r = 0; r < 8; ++r) {
        int b = btile * 16 + r + (group << 3);
        float z3 = 0.0f;
        if (valid) z3 = zeta3[(((size_t)g * MC_ + mc) * B_ + b) * OUT_ + n];
        float d3 = fmaxf(csg[r], 0.0f);
        float o  = fmaxf(cmu[r] + sqrtf(d3) * z3, 0.0f);
        // softmax over the 16 class-lanes of this half-wave (invalid lanes excluded)
        float vmax = valid ? o : -3.4e38f;
        #pragma unroll
        for (int off = 8; off > 0; off >>= 1)
            vmax = fmaxf(vmax, __shfl_xor(vmax, off, 16));
        float e = valid ? __expf(o - vmax) : 0.0f;
        float s = e;
        #pragma unroll
        for (int off = 8; off > 0; off >>= 1)
            s += __shfl_xor(s, off, 16);
        if (valid) {
            float p = (e / s) * (1.0f / (float)MC_);
            atomicAdd(&out[(size_t)b * (GMAX * OUT_) + g * OUT_ + n], p);
        }
    }
}

// ---------------------------------------------------------------------------
extern "C" void kernel_launch(void* const* d_in, const int* in_sizes, int n_in,
                              void* d_out, int out_size, void* d_ws, size_t ws_size,
                              hipStream_t stream) {
    const float* x      = (const float*)d_in[0];
    const float* mu1    = (const float*)d_in[1];
    const float* sigma1 = (const float*)d_in[2];
    const float* mu3    = (const float*)d_in[3];
    const float* sigma3 = (const float*)d_in[4];
    const float* zeta1  = (const float*)d_in[5];
    const float* zeta3  = (const float*)d_in[6];
    const int*   nump   = (const int*)d_in[7];
    float* out = (float*)d_out;

    // Workspace carve-up (all offsets 32B-aligned).
    char* ws = (char*)d_ws;
    size_t off = 0;
    float* G1  = (float*)(ws + off);  off += (size_t)GMAX * B_ * H_ * 4;        // 3.277 MB
    float* SD1 = (float*)(ws + off);  off += (size_t)GMAX * B_ * H_ * 4;        // 3.277 MB
    _Float16* xA    = (_Float16*)(ws + off);  off += (size_t)BT * KT1 * 512 * 2;          // 0.49 MB
    _Float16* x2A   = (_Float16*)(ws + off);  off += (size_t)BT * KT1 * 512 * 2;
    _Float16* w1mu  = (_Float16*)(ws + off);  off += (size_t)GMAX * KT1 * HT * 512 * 2;   // 1.54 MB
    _Float16* w1sig = (_Float16*)(ws + off);  off += (size_t)GMAX * KT1 * HT * 512 * 2;
    _Float16* w3mu  = (_Float16*)(ws + off);  off += (size_t)GMAX * KT3 * 512 * 2;        // 53 KB
    _Float16* w3sig = (_Float16*)(ws + off);  off += (size_t)GMAX * KT3 * 512 * 2;

    // 1) zero output accumulator
    zero_out_kernel<<<(out_size + 255) / 256, 256, 0, stream>>>(out, out_size);

    // 2) pack operands into WMMA lane layouts
    {
        int n = BT * KT1 * 512;
        pack_x_kernel<<<(n + 255) / 256, 256, 0, stream>>>(x, xA, x2A);
    }
    {
        int n = GMAX * KT1 * HT * 512;
        pack_w1_kernel<<<(n + 255) / 256, 256, 0, stream>>>(mu1, sigma1, w1mu, w1sig);
    }
    {
        int n = GMAX * KT3 * 512;
        pack_w3_kernel<<<(n + 255) / 256, 256, 0, stream>>>(mu3, sigma3, w3mu, w3sig);
    }

    // 3) layer-1 GEMMs (shared across all MC samples)
    layer1_wmma_kernel<<<dim3(BT, HT, GMAX), 32, 0, stream>>>(xA, x2A, w1mu, w1sig, G1, SD1, nump);

    // 4) layer-2: stream zeta1 once, WMMA-contract, softmax, MC-mean accumulate
    layer2_wmma_kernel<<<dim3(BT, MC_ / 4, GMAX), 128, 0, stream>>>(
        G1, SD1, w3mu, w3sig, zeta1, zeta3, out, nump);
}
